// KeyNet_79800492359816
// MI455X (gfx1250) — matrix operand order repaired and checked by reading, not compiled, not run on hardware
//
#include <hip/hip_runtime.h>
#include <math.h>

#define NPTS   1000
#define NPAD   1024          // 32 tiles of 32 points
#define NANC   125
#define NTILES 32
#define IMHW   (120 * 160)

typedef __attribute__((ext_vector_type(16))) __bf16 bf16x16;
typedef __attribute__((ext_vector_type(8)))  float  floatx8;

__device__ __forceinline__ unsigned short f2bf(float f) {
    unsigned u = __float_as_uint(f);
    unsigned r = (u + 0x7FFFu + ((u >> 16) & 1u)) >> 16;   // round-to-nearest-even
    return (unsigned short)r;
}

union ABFrag { bf16x16 v; unsigned u[8]; };

// One K-step (K=32): load A (16x32 weights) once, multiply against TWO 32x16
// activation column-tiles, accumulating into two f32 fragments.
__device__ __forceinline__ void wmma_bf16_step2(
    const unsigned short* __restrict__ wtile, int Kpad,
    const unsigned short* __restrict__ x0,    // tile0 LDS base + krow*16
    const unsigned short* __restrict__ x1,    // tile1 LDS base + krow*16
    floatx8& acc0, floatx8& acc1)
{
    const int lane = threadIdx.x & 31;
    const int mrow = lane & 15;
    const int h    = lane >> 4;
    ABFrag A, B0, B1;
    const unsigned short* wr = wtile + (size_t)mrow * Kpad;
#pragma unroll
    for (int r = 0; r < 8; ++r) {
        // 16-bit A 16x32 layout: lanes0-15 regs0-3 -> K0..7, regs4-7 -> K16..23;
        // lanes16-31 shifted by +8 within each half.
        int kb = ((r < 4) ? 0 : 16) + h * 8 + (r & 3) * 2;
        A.u[r] = *(const unsigned*)(wr + kb);
    }
    const unsigned short* xr0 = x0 + mrow;   // lane%16 = N column; K in A's table order
    const unsigned short* xr1 = x1 + mrow;
#pragma unroll
    for (int r = 0; r < 8; ++r) {
        int kb = ((r < 4) ? 0 : 16) + h * 8 + (r & 3) * 2;
        B0.u[r] = (unsigned)xr0[kb * 16] | ((unsigned)xr0[(kb + 1) * 16] << 16);
        B1.u[r] = (unsigned)xr1[kb * 16] | ((unsigned)xr1[(kb + 1) * 16] << 16);
    }
    acc0 = __builtin_amdgcn_wmma_f32_16x16x32_bf16(false, A.v, false, B0.v, (short)0, acc0, false, false);
    acc1 = __builtin_amdgcn_wmma_f32_16x16x32_bf16(false, A.v, false, B1.v, (short)0, acc1, false, false);
}

// ---------------- weight pre-pack (f32 -> bf16, with all_conv1 column permute) ----
__global__ void keynet_pack(const float* __restrict__ W2, const float* __restrict__ W5,
                            const float* __restrict__ Wa1, const float* __restrict__ Wa2,
                            unsigned short* __restrict__ W2p, unsigned short* __restrict__ W5hp,
                            unsigned short* __restrict__ Wa1p, unsigned short* __restrict__ Wa2p)
{
    int idx = blockIdx.x * blockDim.x + threadIdx.x;
    const int n2 = 128 * 64, n5 = 256 * 128, na1 = 320 * 448, na2 = 160 * 320;
    if (idx < n2) { W2p[idx] = f2bf(W2[idx]); return; }
    idx -= n2;
    if (idx < n5) { int m = idx / 128, k = idx % 128; W5hp[idx] = f2bf(W5[m * 256 + k]); return; }
    idx -= n5;
    if (idx < na1) {
        int m = idx / 448, kk = idx % 448;
        int sk = (kk < 64) ? kk : ((kk < 192) ? kk + 64 : kk + 192);  // h1 | h2 | h5 columns
        Wa1p[idx] = f2bf(Wa1[m * 640 + sk]); return;
    }
    idx -= na1;
    if (idx < na2) { Wa2p[idx] = f2bf(Wa2[idx]); }
}

// ---------------- anchor-invariant e-branch precompute (f32 scalar, tiny) ---------
__global__ void keynet_e1(const float* __restrict__ img, const int* __restrict__ choose,
                          const float* __restrict__ We1, const float* __restrict__ be1,
                          float* __restrict__ e1)
{
    int idx = blockIdx.x * blockDim.x + threadIdx.x;
    if (idx >= 64 * NPAD) return;
    int m = idx / NPAD, n = idx % NPAD;
    float v = 0.f;
    if (n < NPTS) {
        int px = choose[n];
        v = be1[m];
        for (int c = 0; c < 32; ++c) v += We1[m * 32 + c] * img[c * IMHW + px];
        v = fmaxf(v, 0.f);
    }
    e1[idx] = v;
}

__global__ void keynet_e2(const float* __restrict__ e1, const float* __restrict__ We2,
                          const float* __restrict__ be2, float* __restrict__ e2)
{
    int idx = blockIdx.x * blockDim.x + threadIdx.x;
    if (idx >= 128 * NPAD) return;
    int m = idx / NPAD, n = idx % NPAD;
    float v = 0.f;
    if (n < NPTS) {
        v = be2[m];
        for (int k = 0; k < 64; ++k) v += We2[m * 64 + k] * e1[k * NPAD + n];
        v = fmaxf(v, 0.f);
    }
    e2[idx] = v;
}

__global__ void keynet_c5e(const float* __restrict__ e2, const float* __restrict__ W5,
                           float* __restrict__ c5e)
{
    int idx = blockIdx.x * blockDim.x + threadIdx.x;
    if (idx >= 256 * NPAD) return;
    int m = idx / NPAD, n = idx % NPAD;
    float v = 0.f;
    if (n < NPTS)
        for (int k = 0; k < 128; ++k) v += W5[m * 256 + 128 + k] * e2[k * NPAD + n];
    c5e[idx] = v;
}

__global__ void keynet_a1e(const float* __restrict__ e1, const float* __restrict__ e2,
                           const float* __restrict__ Wa1, float* __restrict__ a1e)
{
    int idx = blockIdx.x * blockDim.x + threadIdx.x;
    if (idx >= 320 * NPAD) return;
    int m = idx / NPAD, n = idx % NPAD;
    float v = 0.f;
    if (n < NPTS) {
        for (int k = 0; k < 64; ++k)  v += Wa1[m * 640 + 64  + k] * e1[k * NPAD + n];
        for (int k = 0; k < 128; ++k) v += Wa1[m * 640 + 256 + k] * e2[k * NPAD + n];
    }
    a1e[idx] = v;
}

// ---------------- per-anchor distance softmax weights + pooled init ---------------
__global__ __launch_bounds__(128) void keynet_wt(const float* __restrict__ x,
                                                 const float* __restrict__ anchor,
                                                 float* __restrict__ wt,
                                                 float* __restrict__ pooled)
{
    __shared__ float dls[NPTS];
    __shared__ float red[128];
    const int a = blockIdx.x, tid = threadIdx.x;
    const float ax = anchor[a * 3], ay = anchor[a * 3 + 1], az = anchor[a * 3 + 2];
    float lmin = 3.4e38f;
    for (int n = tid; n < NPTS; n += 128) {
        float dx = x[n * 3] - ax, dy = x[n * 3 + 1] - ay, dz = x[n * 3 + 2] - az;
        float d = sqrtf(dx * dx + dy * dy + dz * dz);
        dls[n] = d; lmin = fminf(lmin, d);
    }
    red[tid] = lmin; __syncthreads();
    for (int s = 64; s > 0; s >>= 1) { if (tid < s) red[tid] = fminf(red[tid], red[tid + s]); __syncthreads(); }
    const float dmin = red[0]; __syncthreads();
    float lsum = 0.f;
    for (int n = tid; n < NPTS; n += 128) lsum += expf(dmin - dls[n]);
    red[tid] = lsum; __syncthreads();
    for (int s = 64; s > 0; s >>= 1) { if (tid < s) red[tid] += red[tid + s]; __syncthreads(); }
    const float inv = 1.f / red[0];
    for (int n = tid; n < NPTS; n += 128) wt[(size_t)a * NPAD + n] = expf(dmin - dls[n]) * inv;
    if (tid < NPAD - NPTS) wt[(size_t)a * NPAD + NPTS + tid] = 0.f;
    for (int m = tid; m < 160; m += 128) pooled[a * 160 + m] = 0.f;   // zero accumulators
}

// ---------------- fused per-(anchor, 32-point-tile) MLP + weighted pooling --------
__global__ __launch_bounds__(128) void keynet_main(
    const float* __restrict__ x, const float* __restrict__ anchor,
    const float* __restrict__ W1, const float* __restrict__ b1,
    const unsigned short* __restrict__ W2p, const float* __restrict__ b2,
    const unsigned short* __restrict__ W5hp, const float* __restrict__ b5,
    const float* __restrict__ c5e,
    const unsigned short* __restrict__ Wa1p, const float* __restrict__ ba1,
    const float* __restrict__ a1e,
    const unsigned short* __restrict__ Wa2p, const float* __restrict__ ba2,
    const float* __restrict__ wt, float* __restrict__ pooled)
{
    // [h1(0:64) | h2(64:192) | h5(192:448)] x 16 pts, two column tiles
    __shared__ unsigned short hbuf[2][448 * 16];
    __shared__ unsigned short abuf[2][320 * 16];
    __shared__ float xrls[2][48];
    __shared__ float wtls[32];

    const int a   = blockIdx.x;
    const int t   = blockIdx.y;
    const int tid = threadIdx.x;
    const int wv  = tid >> 5;          // wave id 0..3
    const int lane = tid & 31;
    const int hh   = lane >> 4;        // C-frag half
    const int cn   = lane & 15;        // column within 16-wide tile
    const int ng0  = t * 32 + cn;      // global point, tile0 (arrays are NPAD wide)
    const int ng1  = ng0 + 16;         // global point, tile1

    if (tid < 96) {
        int c = tid % 3, i = tid / 3;          // i = 0..31
        int n = t * 32 + i;
        xrls[i >> 4][c * 16 + (i & 15)] =
            (n < NPTS) ? (x[n * 3 + c] - anchor[a * 3 + c]) : 0.f;
    }
    if (tid < 32) wtls[tid] = wt[(size_t)a * NPAD + t * 32 + tid];
    __syncthreads();

    // conv1: h1 = relu(W1[64x3] @ xr + b1)  (K=3, scalar)
    for (int o = tid; o < 64 * 32; o += 128) {
        int m = o >> 5, n = o & 31;
        int bsel = n >> 4, col = n & 15;
        float v = b1[m];
        v += W1[m * 3 + 0] * xrls[bsel][0 * 16 + col];
        v += W1[m * 3 + 1] * xrls[bsel][1 * 16 + col];
        v += W1[m * 3 + 2] * xrls[bsel][2 * 16 + col];
        hbuf[bsel][m * 16 + col] = f2bf(fmaxf(v, 0.f));
    }
    __syncthreads();

    // conv2: h2 = relu(W2[128x64] @ h1 + b2) -> rows 64..191
    for (int j = 0; j < 2; ++j) {
        int Mb = (wv * 2 + j) * 16;
        floatx8 a0 = {0.f,0.f,0.f,0.f,0.f,0.f,0.f,0.f};
        floatx8 a1 = {0.f,0.f,0.f,0.f,0.f,0.f,0.f,0.f};
        for (int ks = 0; ks < 2; ++ks)
            wmma_bf16_step2(W2p + (size_t)Mb * 64 + ks * 32, 64,
                            hbuf[0] + (ks * 32) * 16, hbuf[1] + (ks * 32) * 16, a0, a1);
#pragma unroll
        for (int r = 0; r < 8; ++r) {
            int gm = Mb + r + 8 * hh;
            hbuf[0][(64 + gm) * 16 + cn] = f2bf(fmaxf(a0[r] + b2[gm], 0.f));
            hbuf[1][(64 + gm) * 16 + cn] = f2bf(fmaxf(a1[r] + b2[gm], 0.f));
        }
    }
    __syncthreads();

    // conv5 (h part): h5 = relu(W5h[256x128] @ h2 + c5e + b5) -> rows 192..447
    for (int j = 0; j < 4; ++j) {
        int Mb = (wv * 4 + j) * 16;
        floatx8 a0 = {0.f,0.f,0.f,0.f,0.f,0.f,0.f,0.f};
        floatx8 a1 = {0.f,0.f,0.f,0.f,0.f,0.f,0.f,0.f};
        for (int ks = 0; ks < 4; ++ks)
            wmma_bf16_step2(W5hp + (size_t)Mb * 128 + ks * 32, 128,
                            hbuf[0] + (64 + ks * 32) * 16, hbuf[1] + (64 + ks * 32) * 16, a0, a1);
#pragma unroll
        for (int r = 0; r < 8; ++r) {
            int gm = Mb + r + 8 * hh;
            float v0 = a0[r] + b5[gm] + c5e[(size_t)gm * NPAD + ng0];
            float v1 = a1[r] + b5[gm] + c5e[(size_t)gm * NPAD + ng1];
            hbuf[0][(192 + gm) * 16 + cn] = f2bf(fmaxf(v0, 0.f));
            hbuf[1][(192 + gm) * 16 + cn] = f2bf(fmaxf(v1, 0.f));
        }
    }
    __syncthreads();

    // all_conv1 (h part): a1 = leaky(Wa1h[320x448] @ [h1;h2;h5] + a1e + ba1)
    for (int j = 0; j < 5; ++j) {
        int Mb = (wv * 5 + j) * 16;
        floatx8 a0 = {0.f,0.f,0.f,0.f,0.f,0.f,0.f,0.f};
        floatx8 a1 = {0.f,0.f,0.f,0.f,0.f,0.f,0.f,0.f};
        for (int ks = 0; ks < 14; ++ks)
            wmma_bf16_step2(Wa1p + (size_t)Mb * 448 + ks * 32, 448,
                            hbuf[0] + (ks * 32) * 16, hbuf[1] + (ks * 32) * 16, a0, a1);
#pragma unroll
        for (int r = 0; r < 8; ++r) {
            int gm = Mb + r + 8 * hh;
            float v0 = a0[r] + ba1[gm] + a1e[(size_t)gm * NPAD + ng0];
            float v1 = a1[r] + ba1[gm] + a1e[(size_t)gm * NPAD + ng1];
            v0 = (v0 > 0.f) ? v0 : 0.01f * v0;
            v1 = (v1 > 0.f) ? v1 : 0.01f * v1;
            abuf[0][gm * 16 + cn] = f2bf(v0);
            abuf[1][gm * 16 + cn] = f2bf(v1);
        }
    }
    __syncthreads();

    // all_conv2: f = Wa2[160x320] @ a1 + ba2 ; pooled[a] += f * wt
    for (int j = 0; j < 3; ++j) {
        int mt = wv + 4 * j;
        if (mt < 10) {   // wave-uniform => EXEC all-ones inside (WMMA requirement)
            int Mb = mt * 16;
            floatx8 a0 = {0.f,0.f,0.f,0.f,0.f,0.f,0.f,0.f};
            floatx8 a1 = {0.f,0.f,0.f,0.f,0.f,0.f,0.f,0.f};
            for (int ks = 0; ks < 10; ++ks)
                wmma_bf16_step2(Wa2p + (size_t)Mb * 320 + ks * 32, 320,
                                abuf[0] + (ks * 32) * 16, abuf[1] + (ks * 32) * 16, a0, a1);
            float w0 = wtls[cn], w1 = wtls[16 + cn];
#pragma unroll
            for (int r = 0; r < 8; ++r) {
                int gm = Mb + r + 8 * hh;
                float bb = ba2[gm];
                float p = (a0[r] + bb) * w0 + (a1[r] + bb) * w1;
                p += __shfl_xor(p, 1);
                p += __shfl_xor(p, 2);
                p += __shfl_xor(p, 4);
                p += __shfl_xor(p, 8);
                if (cn == 0) atomicAdd(&pooled[a * 160 + gm], p);
            }
        }
    }
}

// ---------------- kp / att heads on pooled [125 x 160] -----------------------------
__global__ __launch_bounds__(128) void keynet_head(
    const float* __restrict__ pooled,
    const float* __restrict__ Wkp1, const float* __restrict__ bkp1,
    const float* __restrict__ Wkp2, const float* __restrict__ bkp2,
    const float* __restrict__ Wat1, const float* __restrict__ bat1,
    const float* __restrict__ Wat2, const float* __restrict__ bat2,
    float* __restrict__ kp_out, float* __restrict__ att_out)
{
    __shared__ float feat[160];
    __shared__ float k1[90];
    __shared__ float t1[90];
    const int a = blockIdx.x, tid = threadIdx.x;
    for (int m = tid; m < 160; m += 128) feat[m] = pooled[a * 160 + m];
    __syncthreads();
    for (int m = tid; m < 90; m += 128) {
        float v = bkp1[m], u = bat1[m];
        for (int k = 0; k < 160; ++k) {
            v += Wkp1[m * 160 + k] * feat[k];
            u += Wat1[m * 160 + k] * feat[k];
        }
        k1[m] = (v > 0.f) ? v : 0.01f * v;
        t1[m] = (u > 0.f) ? u : 0.01f * u;
    }
    __syncthreads();
    for (int m = tid; m < 24; m += 128) {
        float v = bkp2[m];
        for (int k = 0; k < 90; ++k) v += Wkp2[m * 90 + k] * k1[k];
        kp_out[a * 24 + m] = v;
    }
    if (tid == 0) {
        float v = bat2[0];
        for (int k = 0; k < 90; ++k) v += Wat2[k] * t1[k];
        att_out[a] = v;
    }
}

// ---------------- att softmax, anchor scaling, argmin, kp select -------------------
__global__ __launch_bounds__(128) void keynet_final(
    const float* __restrict__ anchor, const float* __restrict__ scale,
    const float* __restrict__ gt, const float* __restrict__ att,
    const float* __restrict__ kp_out, float* __restrict__ out)
{
    __shared__ float red[128];
    __shared__ int  ired[128];
    const int tid = threadIdx.x;
    for (int i = tid; i < 375; i += 128) out[24 + i] = anchor[i] * scale[i % 3];

    float lm = -3.4e38f;
    for (int a = tid; a < NANC; a += 128) lm = fmaxf(lm, att[a]);
    red[tid] = lm; __syncthreads();
    for (int s = 64; s > 0; s >>= 1) { if (tid < s) red[tid] = fmaxf(red[tid], red[tid + s]); __syncthreads(); }
    const float amax = red[0]; __syncthreads();
    float ls = 0.f;
    for (int a = tid; a < NANC; a += 128) ls += expf(att[a] - amax);
    red[tid] = ls; __syncthreads();
    for (int s = 64; s > 0; s >>= 1) { if (tid < s) red[tid] += red[tid + s]; __syncthreads(); }
    const float inv = 1.f / red[0]; __syncthreads();
    for (int a = tid; a < NANC; a += 128) out[399 + a] = expf(att[a] - amax) * inv;

    float bd = 3.4e38f; int bi = 0x7fffffff;
    for (int a = tid; a < NANC; a += 128) {
        float d2 = 0.f;
        for (int c = 0; c < 3; ++c) { float dd = anchor[a * 3 + c] * scale[c] - gt[c]; d2 += dd * dd; }
        if (d2 < bd) { bd = d2; bi = a; }
    }
    red[tid] = bd; ired[tid] = bi; __syncthreads();
    for (int s = 64; s > 0; s >>= 1) {
        if (tid < s) {
            bool take = (red[tid + s] < red[tid]) ||
                        (red[tid + s] == red[tid] && ired[tid + s] < ired[tid]);
            if (take) { red[tid] = red[tid + s]; ired[tid] = ired[tid + s]; }
        }
        __syncthreads();
    }
    const int mc = ired[0];
    for (int j = tid; j < 24; j += 128) {
        int c = j % 3;
        out[j] = (kp_out[mc * 24 + j] + anchor[mc * 3 + c]) * scale[c];
    }
}

extern "C" void kernel_launch(void* const* d_in, const int* in_sizes, int n_in,
                              void* d_out, int out_size, void* d_ws, size_t ws_size,
                              hipStream_t stream)
{
    const float* out_img = (const float*)d_in[0];
    const int*   choose  = (const int*)  d_in[1];
    const float* x       = (const float*)d_in[2];
    const float* anchor  = (const float*)d_in[3];
    const float* scale   = (const float*)d_in[4];
    const float* gt_t    = (const float*)d_in[5];
    const float* W1  = (const float*)d_in[6],  *b1  = (const float*)d_in[7];
    const float* We1 = (const float*)d_in[8],  *be1 = (const float*)d_in[9];
    const float* W2  = (const float*)d_in[10], *b2  = (const float*)d_in[11];
    const float* We2 = (const float*)d_in[12], *be2 = (const float*)d_in[13];
    const float* W5  = (const float*)d_in[14], *b5  = (const float*)d_in[15];
    const float* Wa1 = (const float*)d_in[16], *ba1 = (const float*)d_in[17];
    const float* Wa2 = (const float*)d_in[18], *ba2 = (const float*)d_in[19];
    const float* Wk1 = (const float*)d_in[20], *bk1 = (const float*)d_in[21];
    const float* Wk2 = (const float*)d_in[22], *bk2 = (const float*)d_in[23];
    const float* Wt1 = (const float*)d_in[24], *bt1 = (const float*)d_in[25];
    const float* Wt2 = (const float*)d_in[26], *bt2 = (const float*)d_in[27];
    float* out = (float*)d_out;

    char* ws = (char*)d_ws;
    size_t off = 0;
    auto wsalloc = [&](size_t bytes) -> void* {
        void* p = ws + off;
        off = (off + bytes + 255) & ~(size_t)255;
        return p;
    };
    unsigned short* W2p  = (unsigned short*)wsalloc(128 * 64  * 2);
    unsigned short* W5hp = (unsigned short*)wsalloc(256 * 128 * 2);
    unsigned short* Wa1p = (unsigned short*)wsalloc(320 * 448 * 2);
    unsigned short* Wa2p = (unsigned short*)wsalloc(160 * 320 * 2);
    float* e1     = (float*)wsalloc((size_t)64  * NPAD * 4);
    float* e2     = (float*)wsalloc((size_t)128 * NPAD * 4);
    float* c5e    = (float*)wsalloc((size_t)256 * NPAD * 4);
    float* a1e    = (float*)wsalloc((size_t)320 * NPAD * 4);
    float* wt     = (float*)wsalloc((size_t)NANC * NPAD * 4);
    float* pooled = (float*)wsalloc((size_t)NANC * 160 * 4);
    float* kp_out = (float*)wsalloc((size_t)NANC * 24 * 4);
    float* att_o  = (float*)wsalloc((size_t)NANC * 4);

    keynet_pack<<<(235520 + 255) / 256, 256, 0, stream>>>(W2, W5, Wa1, Wa2,
                                                          W2p, W5hp, Wa1p, Wa2p);
    keynet_e1 <<<(64  * NPAD + 127) / 128, 128, 0, stream>>>(out_img, choose, We1, be1, e1);
    keynet_e2 <<<(128 * NPAD + 127) / 128, 128, 0, stream>>>(e1, We2, be2, e2);
    keynet_c5e<<<(256 * NPAD + 127) / 128, 128, 0, stream>>>(e2, W5, c5e);
    keynet_a1e<<<(320 * NPAD + 127) / 128, 128, 0, stream>>>(e1, e2, Wa1, a1e);
    keynet_wt <<<NANC, 128, 0, stream>>>(x, anchor, wt, pooled);

    dim3 mgrid(NANC, NTILES);
    keynet_main<<<mgrid, 128, 0, stream>>>(x, anchor, W1, b1, W2p, b2,
                                           W5hp, b5, c5e, Wa1p, ba1, a1e,
                                           Wa2p, ba2, wt, pooled);

    keynet_head<<<NANC, 128, 0, stream>>>(pooled, Wk1, bk1, Wk2, bk2,
                                          Wt1, bt1, Wt2, bt2, kp_out, att_o);
    keynet_final<<<1, 128, 0, stream>>>(anchor, scale, gt_t, att_o, kp_out, out);

    // out_img passthrough: offset = 24 (kp) + 375 (anchors) + 125 (att)
    hipMemcpyAsync(out + 524, d_in[0], (size_t)32 * IMHW * sizeof(float),
                   hipMemcpyDeviceToDevice, stream);
}
